// CSRNet_Dyn_77275051590020
// MI455X (gfx1250) — matrix operand order repaired and checked by reading, not compiled, not run on hardware
//
#include <hip/hip_runtime.h>
#include <hip/hip_bf16.h>
#include <stdint.h>

typedef _Float16 half_t;
typedef __attribute__((ext_vector_type(16))) _Float16 v16h;
typedef __attribute__((ext_vector_type(8)))  float    v8f;
typedef __attribute__((ext_vector_type(4)))  unsigned int v4u;
typedef __attribute__((ext_vector_type(8)))  int          v8i;
typedef __attribute__((ext_vector_type(4)))  int          v4i;

#ifndef __has_builtin
#define __has_builtin(x) 0
#endif
#if __has_builtin(__builtin_amdgcn_tensor_load_to_lds) && __has_builtin(__builtin_amdgcn_s_wait_tensorcnt)
#define USE_TDM 1
#else
#define USE_TDM 0
#endif

// ---------------- problem constants ----------------
constexpr int NBLK = 3;
constexpr int H0 = 384, W0 = 640, HW0 = H0 * W0;   // 245760
constexpr int H1 = 190, W1 = 318, HW1 = H1 * W1;   // 60420
constexpr int H2 = 95,  W2 = 159, HW2 = H2 * W2;   // 15105
constexpr int H3 = 48,  W3 = 80,  HW3 = H3 * W3;   // 3840
constexpr int NT0 = HW0 / 16;                      // 15360 (exact)
constexpr int NT1 = (HW1 + 15) / 16;               // 3777
constexpr int NT2 = (HW2 + 15) / 16;               // 945
constexpr int NT3 = HW3 / 16;                      // 240 (exact)

// A-matrix (16x32 f16) K index for vector element j (matches B fill):
// lanes 0-15: K=0..15 ; lanes 16-31: K=16..31
__device__ __forceinline__ int a_kmap(int lane, int j) {
    return ((lane & 16) ? 16 : 0) + j;
}

__device__ __forceinline__ float sigf(float x) { return 1.0f / (1.0f + __expf(-x)); }

// Escape the LDS pointer into an opaque asm with a memory clobber. The asm may
// have written the LDS (so staging done by TDM / reinterpret-cast stores cannot
// be proven dead and the subsequent reads cannot fold to poison), but the
// pointer used for the reads keeps its __shared__ provenance, so the accesses
// stay in addrspace(3) and lower to ds_load (not flat_load).
__device__ __forceinline__ void lds_publish(const void* p) {
    asm volatile("" : : "v"(p) : "memory");
}

#if USE_TDM
// Issue one TDM 2D tile load (tile_w halves x tile_h rows, row pitch in halves)
// from global memory into LDS at byte offset lds_off. D# packing per ISA 8.3/8.4.
__device__ __forceinline__ void tdm_load_2d_f16(uint32_t lds_off, const void* gaddr,
                                                uint32_t tile_w, uint32_t tile_h,
                                                uint64_t pitch_elems) {
    uint64_t ga = (uint64_t)(uintptr_t)gaddr;
    v4u g0 = {};
    g0[0] = 1u;                                            // count=1 (valid), user mode
    g0[1] = lds_off;                                       // lds_addr
    g0[2] = (uint32_t)(ga & 0xFFFFFFFFu);                  // global_addr[31:0]
    g0[3] = (uint32_t)((ga >> 32) & 0x1FFFFFFu) | (2u << 30); // global_addr[56:32] | type=2
    v8i g1 = {};
    g1[0] = (int)(1u << 16);                               // data_size=1 (2 bytes)
    g1[1] = (int)((tile_w & 0xFFFFu) << 16);               // tensor_dim0[15:0]
    g1[2] = (int)(((tile_w >> 16) & 0xFFFFu) | ((tile_h & 0xFFFFu) << 16)); // dim0 hi | dim1 lo
    g1[3] = (int)(((tile_h >> 16) & 0xFFFFu) | ((tile_w & 0xFFFFu) << 16)); // dim1 hi | tile_dim0
    g1[4] = (int)(tile_h & 0xFFFFu);                       // tile_dim1 (tile_dim2=0)
    g1[5] = (int)(uint32_t)(pitch_elems & 0xFFFFFFFFu);    // tensor_dim0_stride[31:0]
    g1[6] = (int)(uint32_t)((pitch_elems >> 32) & 0xFFFFu);// dim0_stride[47:32] | dim1_stride lo
    g1[7] = 0;
    v4i g2 = {};
    v4i g3 = {};
#if defined(__clang_major__) && (__clang_major__ >= 23)
    v8i g4 = {};
    __builtin_amdgcn_tensor_load_to_lds(g0, g1, g2, g3, g4, 0);
#else
    __builtin_amdgcn_tensor_load_to_lds(g0, g1, g2, g3, 0);
#endif
}
#endif

// ---------------- tiny dense network: rep -> biLSTM -> ow ----------------
__global__ void smallnet_kernel(const float* __restrict__ cond,
                                const float* __restrict__ rep_w1,   // (64,7680)
                                const float* __restrict__ rep_w2,   // (2048,64)
                                const float* __restrict__ lstm_fw_w,// (256,128)
                                const float* __restrict__ lstm_bw_w,// (256,128)
                                const float* __restrict__ lconv_w,  // (6,64,128)
                                const float* __restrict__ lconv_b,  // (6,64)
                                float* h1, float* rep,
                                float* hf, float* cf, float* hb, float* cb,
                                float* gates, float* wsF, float* wsB, float* ow) {
    const int t = threadIdx.x;
    if (t < 128) {
        int b = t >> 6, i = t & 63;
        float acc = 0.f;
        for (int k = 0; k < 7680; ++k) acc += cond[b * 7680 + k] * rep_w1[i * 7680 + k];
        h1[t] = acc > 0.f ? acc : 0.1f * acc;
    }
    __syncthreads();
    for (int idx = t; idx < 4096; idx += 256) {
        int b = idx >> 11, o = idx & 2047;
        float acc = 0.f;
        for (int k = 0; k < 64; ++k) acc += h1[b * 64 + k] * rep_w2[o * 64 + k];
        rep[(b * 32 + (o >> 6)) * 64 + (o & 63)] = acc;
    }
    __syncthreads();
    for (int idx = t; idx < 4096; idx += 256) { hf[idx] = 0.f; cf[idx] = 0.f; hb[idx] = 0.f; cb[idx] = 0.f; }
    __syncthreads();
    for (int s = 0; s < 6; ++s) {
        for (int dir = 0; dir < 2; ++dir) {
            const float* w = dir ? lstm_bw_w : lstm_fw_w;
            float* h = dir ? hb : hf;
            float* c = dir ? cb : cf;
            float* wsd = dir ? wsB : wsF;
            for (int idx = t; idx < 16384; idx += 256) {
                int r = idx >> 8, g = idx & 255;
                float acc = 0.f;
                for (int k = 0; k < 64; ++k) acc += rep[r * 64 + k] * w[g * 128 + k];
                for (int k = 0; k < 64; ++k) acc += h[r * 64 + k] * w[g * 128 + 64 + k];
                gates[idx] = acc;
            }
            __syncthreads();
            for (int idx = t; idx < 4096; idx += 256) {
                int r = idx >> 6, f = idx & 63;
                float gi = gates[r * 256 + f];
                float gf = gates[r * 256 + 64 + f];
                float go = gates[r * 256 + 128 + f];
                float gg = gates[r * 256 + 192 + f];
                float cn = sigf(gf) * c[idx] + sigf(gi) * tanhf(gg);
                float hn = sigf(go) * tanhf(cn);
                c[idx] = cn; h[idx] = hn;
                wsd[s * 4096 + idx] = hn;
            }
            __syncthreads();
        }
    }
    for (int idx = t; idx < 6 * 4096; idx += 256) {
        int s = idx >> 12, rem = idx & 4095, r = rem >> 6, f = rem & 63;
        float acc = lconv_b[s * 64 + f];
        const float* w = lconv_w + (s * 64 + f) * 128;
        for (int k = 0; k < 64; ++k) acc += wsF[s * 4096 + r * 64 + k] * w[k];
        for (int k = 0; k < 64; ++k) acc += wsB[(5 - s) * 4096 + r * 64 + k] * w[64 + k];
        ow[idx] = acc;
    }
}

// ---------------- conv_first: 1x1 3->64, write f32 + f16 mirror ----------------
__global__ void conv_first_kernel(const float* __restrict__ x,
                                  const float* __restrict__ w,   // (64,3)
                                  const float* __restrict__ bias,
                                  float* __restrict__ x1F, half_t* __restrict__ x1H) {
    int idx = blockIdx.x * 256 + threadIdx.x;
    if (idx >= 2 * HW0) return;
    int b = idx / HW0, p = idx - b * HW0;
    float a0 = x[(b * 3 + 0) * HW0 + p];
    float a1 = x[(b * 3 + 1) * HW0 + p];
    float a2 = x[(b * 3 + 2) * HW0 + p];
    for (int o = 0; o < 64; ++o) {
        float v = w[o * 3] * a0 + w[o * 3 + 1] * a1 + w[o * 3 + 2] * a2 + bias[o];
        int oi = (b * 64 + o) * HW0 + p;
        x1F[oi] = v; x1H[oi] = (half_t)v;
    }
}

// ---------------- pack conv weights (32,Ktot) -> CDNA5 A layout ----------------
__global__ void pack_conv_kernel(const float* __restrict__ w, int ksteps, half_t* __restrict__ pA) {
    int idx = blockIdx.x * 256 + threadIdx.x;
    int total = 2 * ksteps * 512;
    if (idx >= total) return;
    int j = idx & 15, lane = (idx >> 4) & 31;
    int mk = idx >> 9, ks = mk % ksteps, mt = mk / ksteps;
    int m = mt * 16 + (lane & 15);
    int k = ks * 32 + a_kmap(lane, j);
    pA[idx] = (half_t)w[m * (ksteps * 32) + k];
}

// ---------------- pack fused dyn/share 64x64 weights per batch ----------------
__global__ void pack_dyn_kernel(const float* __restrict__ dynw,   // (32,64)
                                const float* __restrict__ sharew, // (32,64)
                                const float* __restrict__ ow_s,   // (64,64) = (b,32,64)
                                half_t* __restrict__ pW) {
    int idx = blockIdx.x * 256 + threadIdx.x;            // 8192 total
    if (idx >= 8192) return;
    int j = idx & 15, lane = (idx >> 4) & 31;
    int ks = (idx >> 9) & 1, mt = (idx >> 10) & 3, b = idx >> 12;
    int m = mt * 16 + (lane & 15);
    int k = ks * 32 + a_kmap(lane, j);
    float v = (m < 32) ? dynw[m * 64 + k] * ow_s[(b * 32 + m) * 64 + k]
                       : sharew[(m - 32) * 64 + k];
    pW[idx] = (half_t)v;
}

// ---------------- WMMA conv: stride2 pad1, 32 out-ch, relu+bias, f16/f32 out ----------------
template <int KH, int KW, int CIN>
__global__ void conv_wmma_kernel(const half_t* __restrict__ src, int Hin, int Win,
                                 const half_t* __restrict__ pA, const float* __restrict__ bias,
                                 half_t* __restrict__ dstH, float* __restrict__ dstF,
                                 int Hout, int Wout, int ntiles) {
    constexpr int KHKW = KH * KW;
    constexpr int KSTEPS = CIN * KH * KW / 32;
    int wave = blockIdx.x * 8 + (threadIdx.x >> 5);
    int lane = threadIdx.x & 31;
    int total = 2 * 2 * ntiles;
    if (wave >= total) return;                       // wave-uniform
    int b = wave / (2 * ntiles);
    int rem = wave - b * 2 * ntiles;
    int mt = rem / ntiles, nt = rem - mt * ntiles;
    int HWo = Hout * Wout;
    int n = nt * 16 + (lane & 15);
    bool nvalid = n < HWo;
    int oh = n / Wout, owc = n - oh * Wout;
    int hbase = oh * 2 - 1, wbase = owc * 2 - 1;
    const half_t* sb = src + (size_t)b * CIN * Hin * Win;
    int kb = (lane & 16) ? 16 : 0;                   // B layout: lanes 16-31 hold K+16
    v8f acc = {};
    for (int ks = 0; ks < KSTEPS; ++ks) {
        v16h a = *(const v16h*)(pA + (size_t)((mt * KSTEPS + ks) * 32 + lane) * 16);
        v16h bv;
#pragma unroll
        for (int j = 0; j < 16; ++j) {
            int k = ks * 32 + kb + j;
            int c = k / KHKW;
            int rr = k - c * KHKW;
            int kh = rr / KW, kw = rr - kh * KW;
            int hin = hbase + kh, win = wbase + kw;
            half_t v = (half_t)0.f;
            if (nvalid && hin >= 0 && hin < Hin && win >= 0 && win < Win)
                v = sb[(size_t)(c * Hin + hin) * Win + win];
            bv[j] = v;
        }
        acc = __builtin_amdgcn_wmma_f32_16x16x32_f16(false, a, false, bv, (short)0, acc, false, false);
    }
    if (nvalid) {
        int mofs = (lane & 16) ? 8 : 0;
#pragma unroll
        for (int v = 0; v < 8; ++v) {
            int m = mt * 16 + v + mofs;
            float tv = acc[v] + bias[m];
            tv = tv > 0.f ? tv : 0.f;                // all condition convs use relu
            size_t oi = (size_t)(b * 32 + m) * HWo + n;
            if (dstH) dstH[oi] = (half_t)tv;
            if (dstF) dstF[oi] = tv;
        }
    }
}

// ---------------- shared LDS staging for pointwise GEMMs ----------------
// Block = 8 waves = 4 M-tiles x 2 N-tiles over one (b, 32-pixel) slab.
// Slab (64 ch x 32 px f16 = 4KB) staged once: TDM async tensor load if
// available, else coalesced b128 loads. lds_publish() makes the staging
// visible to alias analysis while keeping addrspace(3) provenance.
__device__ __forceinline__ void stage_slab(half_t* smem, const half_t* gsrc) {
#if USE_TDM
    if ((threadIdx.x >> 5) == 0) {
        tdm_load_2d_f16(0u, gsrc, 32u, 64u, (uint64_t)HW0);
        __builtin_amdgcn_s_wait_tensorcnt(0);
    }
    __syncthreads();
#else
    int t = threadIdx.x;
    int row = t >> 2, seg = t & 3;                    // 64 rows x 4 x 16B
    const uint4* s = (const uint4*)(gsrc + (size_t)row * HW0 + seg * 8);
    *(uint4*)(&smem[row * 32 + seg * 8]) = *s;
    __syncthreads();
#endif
    lds_publish(smem);
}

// ---------------- WMMA pointwise stage1: out = relu((Wc@x1 + bias)*(1+scale)+shift) -> f16 ----------------
__global__ void pw_stage1_kernel(const half_t* __restrict__ x1H, const half_t* __restrict__ pW,
                                 const float* __restrict__ share_b, const float* __restrict__ scaleb,
                                 const float* __restrict__ shiftb, half_t* __restrict__ outH) {
    __shared__ half_t smem[64 * 32];
    int b = blockIdx.x / (NT0 / 2);
    int ntp = blockIdx.x - b * (NT0 / 2);
    int nbase = ntp * 32;
    stage_slab(smem, x1H + (size_t)b * 64 * HW0 + nbase);

    int wv = threadIdx.x >> 5, lane = threadIdx.x & 31;
    int mt = wv & 3, nh = wv >> 2;
    int col = nh * 16 + (lane & 15);
    int kb = (lane & 16) ? 16 : 0;
    v8f acc = {};
#pragma unroll
    for (int ks = 0; ks < 2; ++ks) {
        v16h a = *(const v16h*)(pW + (size_t)(((b * 4 + mt) * 2 + ks) * 32 + lane) * 16);
        v16h bv;
#pragma unroll
        for (int j = 0; j < 16; ++j) bv[j] = smem[(ks * 32 + kb + j) * 32 + col];
        acc = __builtin_amdgcn_wmma_f32_16x16x32_f16(false, a, false, bv, (short)0, acc, false, false);
    }
    int n = nbase + col;
    int mofs = (lane & 16) ? 8 : 0;
#pragma unroll
    for (int v = 0; v < 8; ++v) {
        int m = mt * 16 + v + mofs;
        float tv = acc[v] + (m >= 32 ? share_b[m - 32] : 0.f);
        tv = tv * (1.f + scaleb[b * 64 + m]) + shiftb[b * 64 + m];
        tv = tv > 0.f ? tv : 0.f;
        outH[(size_t)(b * 64 + m) * HW0 + n] = (half_t)tv;
    }
}

// ---------------- WMMA pointwise stage2: x1 += Wc@out + bias ; refresh f16 mirror ----------------
__global__ void pw_stage2_kernel(const half_t* __restrict__ srcH, const half_t* __restrict__ pW,
                                 const float* __restrict__ share_b,
                                 float* __restrict__ x1F, half_t* __restrict__ x1H) {
    __shared__ half_t smem[64 * 32];
    int b = blockIdx.x / (NT0 / 2);
    int ntp = blockIdx.x - b * (NT0 / 2);
    int nbase = ntp * 32;
    stage_slab(smem, srcH + (size_t)b * 64 * HW0 + nbase);

    int wv = threadIdx.x >> 5, lane = threadIdx.x & 31;
    int mt = wv & 3, nh = wv >> 2;
    int col = nh * 16 + (lane & 15);
    int kb = (lane & 16) ? 16 : 0;
    v8f acc = {};
#pragma unroll
    for (int ks = 0; ks < 2; ++ks) {
        v16h a = *(const v16h*)(pW + (size_t)(((b * 4 + mt) * 2 + ks) * 32 + lane) * 16);
        v16h bv;
#pragma unroll
        for (int j = 0; j < 16; ++j) bv[j] = smem[(ks * 32 + kb + j) * 32 + col];
        acc = __builtin_amdgcn_wmma_f32_16x16x32_f16(false, a, false, bv, (short)0, acc, false, false);
    }
    int n = nbase + col;
    int mofs = (lane & 16) ? 8 : 0;
#pragma unroll
    for (int v = 0; v < 8; ++v) {
        int m = mt * 16 + v + mofs;
        float tv = acc[v] + (m >= 32 ? share_b[m - 32] : 0.f);
        size_t oi = (size_t)(b * 64 + m) * HW0 + n;
        float nv = x1F[oi] + tv;
        x1F[oi] = nv; x1H[oi] = (half_t)nv;
    }
}

// ---------------- mean over HW3 per (b,ch) ----------------
__global__ void reduce_mean_kernel(const float* __restrict__ c3, float* __restrict__ condv) {
    __shared__ float sm[256];
    int row = blockIdx.x;                // b*32+ch, 64 rows
    float s = 0.f;
    for (int i = threadIdx.x; i < HW3; i += 256) s += c3[(size_t)row * HW3 + i];
    sm[threadIdx.x] = s; __syncthreads();
    for (int o = 128; o > 0; o >>= 1) {
        if (threadIdx.x < o) sm[threadIdx.x] += sm[threadIdx.x + o];
        __syncthreads();
    }
    if (threadIdx.x == 0) condv[row] = sm[0] / (float)HW3;
}

// ---------------- scale/shift heads: (2,M) = cond @ W.T + b ----------------
__global__ void affine_cond_kernel(const float* __restrict__ condv,
                                   const float* __restrict__ wsc, const float* __restrict__ bsc,
                                   const float* __restrict__ wsh, const float* __restrict__ bsh,
                                   int M, float* __restrict__ scaleo, float* __restrict__ shifto) {
    int t = threadIdx.x;
    if (t >= 2 * 2 * M) return;
    int which = t / (2 * M);
    int rem = t - which * 2 * M;
    int b = rem / M, m = rem - b * M;
    const float* w = which ? wsh : wsc;
    const float* bb = which ? bsh : bsc;
    float acc = bb[m];
    for (int c = 0; c < 32; ++c) acc += condv[b * 32 + c] * w[m * 32 + c];
    (which ? shifto : scaleo)[b * M + m] = acc;
}

// ---------------- final: out = (fconv@x1+fb)*(1+scale)+shift ----------------
__global__ void final_kernel(const float* __restrict__ x1F, const float* __restrict__ fw,
                             const float* __restrict__ fb, const float* __restrict__ sc,
                             const float* __restrict__ sh, float* __restrict__ out) {
    int idx = blockIdx.x * 256 + threadIdx.x;
    if (idx >= 2 * HW0) return;
    int b = idx / HW0, p = idx - b * HW0;
    float a0 = fb[0], a1 = fb[1];
    for (int c = 0; c < 64; ++c) {
        float xv = x1F[(size_t)(b * 64 + c) * HW0 + p];
        a0 += fw[c] * xv;
        a1 += fw[64 + c] * xv;
    }
    out[(size_t)(b * 2 + 0) * HW0 + p] = a0 * (1.f + sc[b * 2 + 0]) + sh[b * 2 + 0];
    out[(size_t)(b * 2 + 1) * HW0 + p] = a1 * (1.f + sc[b * 2 + 1]) + sh[b * 2 + 1];
}

extern "C" void kernel_launch(void* const* d_in, const int* in_sizes, int n_in,
                              void* d_out, int out_size, void* d_ws, size_t ws_size,
                              hipStream_t stream) {
    (void)in_sizes; (void)n_in; (void)out_size; (void)ws_size;
    const float* x            = (const float*)d_in[0];
    const float* cond         = (const float*)d_in[1];
    const float* rep_w1       = (const float*)d_in[2];
    const float* rep_w2       = (const float*)d_in[3];
    const float* lstm_fw_w    = (const float*)d_in[4];
    const float* lstm_bw_w    = (const float*)d_in[5];
    const float* lstm_conv_w  = (const float*)d_in[6];
    const float* lstm_conv_b  = (const float*)d_in[7];
    const float* conv_first_w = (const float*)d_in[8];
    const float* conv_first_b = (const float*)d_in[9];
    const float* bc1_w        = (const float*)d_in[10];
    const float* bc1_b        = (const float*)d_in[11];
    const float* bc2_w        = (const float*)d_in[12];
    const float* bc2_b        = (const float*)d_in[13];
    const float* bc3_w        = (const float*)d_in[14];
    const float* bc3_b        = (const float*)d_in[15];
    const float* bscale_w     = (const float*)d_in[16];
    const float* bscale_b     = (const float*)d_in[17];
    const float* bshift_w     = (const float*)d_in[18];
    const float* bshift_b     = (const float*)d_in[19];
    const float* bshare1_w    = (const float*)d_in[20];
    const float* bshare1_b    = (const float*)d_in[21];
    const float* bdyn1_w      = (const float*)d_in[22];
    const float* bshare2_w    = (const float*)d_in[23];
    const float* bshare2_b    = (const float*)d_in[24];
    const float* bdyn2_w      = (const float*)d_in[25];
    const float* fc1_w        = (const float*)d_in[26];
    const float* fc1_b        = (const float*)d_in[27];
    const float* fc2_w        = (const float*)d_in[28];
    const float* fc2_b        = (const float*)d_in[29];
    const float* fc3_w        = (const float*)d_in[30];
    const float* fc3_b        = (const float*)d_in[31];
    const float* fconv_w      = (const float*)d_in[32];
    const float* fconv_b      = (const float*)d_in[33];
    const float* fscale_w     = (const float*)d_in[34];
    const float* fscale_b     = (const float*)d_in[35];
    const float* fshift_w     = (const float*)d_in[36];
    const float* fshift_b     = (const float*)d_in[37];

    // ---- workspace carve-up ----
    size_t off = 0;
    auto alloc = [&](size_t bytes) -> void* {
        off = (off + 255) & ~(size_t)255;
        void* p = (char*)d_ws + off;
        off += bytes;
        return p;
    };
    float*  x1F  = (float*) alloc((size_t)2 * 64 * HW0 * 4);
    half_t* x1H  = (half_t*)alloc((size_t)2 * 64 * HW0 * 2);
    half_t* outH = (half_t*)alloc((size_t)2 * 64 * HW0 * 2);
    half_t* c1H  = (half_t*)alloc((size_t)2 * 32 * HW1 * 2);
    half_t* c2H  = (half_t*)alloc((size_t)2 * 32 * HW2 * 2);
    float*  c3F  = (float*) alloc((size_t)2 * 32 * HW3 * 4);
    half_t* pC1  = (half_t*)alloc((size_t)2 * 98 * 512 * 2);
    half_t* pC2  = (half_t*)alloc((size_t)2 * 9 * 512 * 2);
    half_t* pC3  = (half_t*)alloc((size_t)2 * 9 * 512 * 2);
    half_t* pW1  = (half_t*)alloc(8192 * 2);
    half_t* pW2  = (half_t*)alloc(8192 * 2);
    float* h1    = (float*)alloc(128 * 4);
    float* rep   = (float*)alloc(4096 * 4);
    float* hf    = (float*)alloc(4096 * 4);
    float* cf    = (float*)alloc(4096 * 4);
    float* hb    = (float*)alloc(4096 * 4);
    float* cb    = (float*)alloc(4096 * 4);
    float* gates = (float*)alloc(16384 * 4);
    float* wsF   = (float*)alloc(6 * 4096 * 4);
    float* wsB   = (float*)alloc(6 * 4096 * 4);
    float* owb   = (float*)alloc(6 * 4096 * 4);
    float* condv = (float*)alloc(64 * 4);
    float* scaleb= (float*)alloc(128 * 4);
    float* shiftb= (float*)alloc(128 * 4);
    float* sc2   = (float*)alloc(4 * 4);
    float* sh2   = (float*)alloc(4 * 4);

    // ---- small net ----
    smallnet_kernel<<<1, 256, 0, stream>>>(cond, rep_w1, rep_w2, lstm_fw_w, lstm_bw_w,
                                           lstm_conv_w, lstm_conv_b,
                                           h1, rep, hf, cf, hb, cb, gates, wsF, wsB, owb);
    // ---- x1 = conv_first(x) ----
    conv_first_kernel<<<(2 * HW0 + 255) / 256, 256, 0, stream>>>(x, conv_first_w, conv_first_b, x1F, x1H);

    auto cond_stack = [&](const float* w1, const float* b1, const float* w2, const float* b2,
                          const float* w3, const float* b3) {
        pack_conv_kernel<<<(2 * 98 * 512 + 255) / 256, 256, 0, stream>>>(w1, 98, pC1);
        conv_wmma_kernel<7, 7, 64><<<(2 * 2 * NT1 + 7) / 8, 256, 0, stream>>>(
            x1H, H0, W0, pC1, b1, c1H, nullptr, H1, W1, NT1);
        pack_conv_kernel<<<(2 * 9 * 512 + 255) / 256, 256, 0, stream>>>(w2, 9, pC2);
        conv_wmma_kernel<3, 3, 32><<<(2 * 2 * NT2 + 7) / 8, 256, 0, stream>>>(
            c1H, H1, W1, pC2, b2, c2H, nullptr, H2, W2, NT2);
        pack_conv_kernel<<<(2 * 9 * 512 + 255) / 256, 256, 0, stream>>>(w3, 9, pC3);
        conv_wmma_kernel<3, 3, 32><<<(2 * 2 * NT3 + 7) / 8, 256, 0, stream>>>(
            c2H, H2, W2, pC3, b3, nullptr, c3F, H3, W3, NT3);
        reduce_mean_kernel<<<64, 256, 0, stream>>>(c3F, condv);
    };

    const int pwBlocks = 2 * (NT0 / 2);  // 15360 blocks, 8 waves each (4 Mt x 2 Nt)
    for (int j = 0; j < NBLK; ++j) {
        cond_stack(bc1_w + (size_t)j * 32 * 64 * 49, bc1_b + j * 32,
                   bc2_w + (size_t)j * 32 * 32 * 9,  bc2_b + j * 32,
                   bc3_w + (size_t)j * 32 * 32 * 9,  bc3_b + j * 32);
        affine_cond_kernel<<<1, 256, 0, stream>>>(condv,
            bscale_w + (size_t)j * 64 * 32, bscale_b + j * 64,
            bshift_w + (size_t)j * 64 * 32, bshift_b + j * 64, 64, scaleb, shiftb);
        pack_dyn_kernel<<<32, 256, 0, stream>>>(bdyn1_w + (size_t)j * 2048,
                                                bshare1_w + (size_t)j * 2048,
                                                owb + (size_t)(2 * j) * 4096, pW1);
        pw_stage1_kernel<<<pwBlocks, 256, 0, stream>>>(x1H, pW1, bshare1_b + j * 32,
                                                       scaleb, shiftb, outH);
        pack_dyn_kernel<<<32, 256, 0, stream>>>(bdyn2_w + (size_t)j * 2048,
                                                bshare2_w + (size_t)j * 2048,
                                                owb + (size_t)(2 * j + 1) * 4096, pW2);
        pw_stage2_kernel<<<pwBlocks, 256, 0, stream>>>(outH, pW2, bshare2_b + j * 32, x1F, x1H);
    }

    // ---- final head ----
    cond_stack(fc1_w, fc1_b, fc2_w, fc2_b, fc3_w, fc3_b);
    affine_cond_kernel<<<1, 256, 0, stream>>>(condv, fscale_w, fscale_b, fshift_w, fshift_b,
                                              2, sc2, sh2);
    final_kernel<<<(2 * HW0 + 255) / 256, 256, 0, stream>>>(x1F, fconv_w, fconv_b, sc2, sh2,
                                                            (float*)d_out);
}
